// FrameTransformer_18013092839841
// MI455X (gfx1250) — matrix operand order repaired and checked by reading, not compiled, not run on hardware
//
#include <hip/hip_runtime.h>
#include <cstdint>

// ---------------------------------------------------------------------------
// FrameTransformer for MI455X (gfx1250, wave32, WMMA).
// All large GEMMs run through v_wmma_f32_16x16x32_bf16 (bf16 in, f32 accum).
// Workspace arena (assumes ws_size >= ~736MB), with region reuse:
//   [0,64)    x residual fp32
//   [64,96)   xnT bf16   (reused for xn2T)
//   [96,160)  Wq/Wk/Wv/Wo bf16 (16MB each)
//   [160,288) Wl1/Wl2 bf16 (64MB each)
//   [288,384) p_q/p_k/p_v bf16 (32MB each; p_q region reused for aT)
//   [384,480) q/k/vT bf16 (32MB each)
//   [480,736) logits fp32 (attn bf16 aliased in place; later h + h2 bf16)
// ---------------------------------------------------------------------------

#define B_   4
#define CIN_ 2
#define C_   8
#define F_   1024
#define W_   512
#define H_   8
#define DH_  128
#define FE_  4096   // F * EXP

typedef __attribute__((ext_vector_type(16))) __bf16          v16bf;
typedef __attribute__((ext_vector_type(8)))  float           v8f;
typedef __attribute__((ext_vector_type(8)))  unsigned short  uh8;
typedef __attribute__((ext_vector_type(16))) unsigned short  uh16;

__device__ __forceinline__ unsigned short f2bf(float f) {
  unsigned u = __builtin_bit_cast(unsigned, f);
  u += 0x7FFFu + ((u >> 16) & 1u);          // round-to-nearest-even
  return (unsigned short)(u >> 16);
}
__device__ __forceinline__ float bf2f(unsigned short h) {
  return __builtin_bit_cast(float, ((unsigned)h) << 16);
}

// ---------------------------------------------------------------------------
// Generic batched WMMA GEMM:  D[m][n] = sum_k A[m][k] * Bt[n][k]  (+ epilogue)
// A: row-major M x K bf16 (K contiguous), Bt: row-major N x K bf16.
// Per-z offsets: off(z) = (z/div)*s1 + (z%div)*s2   (handles per-c weights and
// the attention-head output interleave into aT).
// Block = 128 threads = 4 waves in 2x2; wave tile 64x64; block tile 128x128.
// All M,N,K in this net are multiples of the tiles -> no bounds checks.
// EPI: 0 = fp32 store, 1 = fp32 store + residual add, 2 = bf16 store.
// ---------------------------------------------------------------------------
template <int EPI>
__global__ __launch_bounds__(128) void gemm_bf16_wmma(
    const unsigned short* __restrict__ A, const unsigned short* __restrict__ Bt,
    float* __restrict__ outF, unsigned short* __restrict__ outH,
    const float* __restrict__ res,
    int lda, int ldb, int ldc, int K,
    int adiv, long long as1, long long as2,
    int bdiv, long long bs1, long long bs2,
    int odiv, long long os1, long long os2)
{
  const int z = blockIdx.z;
  A  += (long long)(z / adiv) * as1 + (long long)(z % adiv) * as2;
  Bt += (long long)(z / bdiv) * bs1 + (long long)(z % bdiv) * bs2;
  const long long ooff = (long long)(z / odiv) * os1 + (long long)(z % odiv) * os2;

  const int lane = threadIdx.x & 31;
  const int wave = threadIdx.x >> 5;
  const int m_w  = blockIdx.x * 128 + (wave >> 1) * 64;
  const int n_w  = blockIdx.y * 128 + (wave & 1) * 64;
  const int l15  = lane & 15;
  const int lhi  = lane >> 4;

  // A fragment (16x32 bf16): lane m holds k-chunks [base..base+7] and
  // [base+16..base+23], base = (lane<16)?0:8  -> two 16B loads per tile.
  // B fragment (32x16 bf16): lane n holds k-chunks [base..base+7] and
  // [base+16..base+23], base = (lane<16)?0:8  -> two 16B loads per tile.
  const unsigned short* arow[4];
  const unsigned short* brow[4];
#pragma unroll
  for (int i = 0; i < 4; ++i) {
    arow[i] = A  + (long long)(m_w + i * 16 + l15) * lda + (lhi << 3);
    brow[i] = Bt + (long long)(n_w + i * 16 + l15) * ldb + (lhi << 3);
  }

  v8f acc[4][4] = {};

  for (int k0 = 0; k0 < K; k0 += 32) {
    v16bf af[4], bfr[4];
#pragma unroll
    for (int i = 0; i < 4; ++i) {
      uh16 t;
      *(uh8*)&t        = *(const uh8*)(arow[i] + k0);
      ((uh8*)&t)[1]    = *(const uh8*)(arow[i] + k0 + 16);
      af[i] = __builtin_bit_cast(v16bf, t);
    }
#pragma unroll
    for (int i = 0; i < 4; ++i) {
      uh16 t;
      *(uh8*)&t        = *(const uh8*)(brow[i] + k0);
      ((uh8*)&t)[1]    = *(const uh8*)(brow[i] + k0 + 16);
      bfr[i] = __builtin_bit_cast(v16bf, t);
    }
#pragma unroll
    for (int im = 0; im < 4; ++im)
#pragma unroll
      for (int in = 0; in < 4; ++in)
        acc[im][in] = __builtin_amdgcn_wmma_f32_16x16x32_bf16(
            false, af[im], false, bfr[in], (short)0, acc[im][in], false, false);
  }

  // D layout: element r of lane l is (m = r + 8*(l>>4), n = l&15) within tile.
#pragma unroll
  for (int im = 0; im < 4; ++im)
#pragma unroll
    for (int in = 0; in < 4; ++in)
#pragma unroll
      for (int r = 0; r < 8; ++r) {
        const int m = m_w + im * 16 + r + (lhi << 3);
        const int n = n_w + in * 16 + l15;
        const long long idx = ooff + (long long)m * ldc + n;
        float v = acc[im][in][r];
        if (EPI == 1) v += res[idx];
        if (EPI == 2) outH[idx] = f2bf(v);
        else          outF[idx] = v;
      }
}

// ---------------------------------------------------------------------------
// fp32 -> bf16 bulk convert (weights staging), 4 elements/thread.
// ---------------------------------------------------------------------------
__global__ __launch_bounds__(256) void k_f32_to_bf16(
    const float* __restrict__ in, unsigned short* __restrict__ out, long long n4)
{
  long long i = (long long)blockIdx.x * 256 + threadIdx.x;
  if (i >= n4) return;
  const float4 v = ((const float4*)in)[i];
  ushort4 o;
  o.x = f2bf(v.x); o.y = f2bf(v.y); o.z = f2bf(v.z); o.w = f2bf(v.w);
  ((ushort4*)out)[i] = o;
}

// ---------------------------------------------------------------------------
// embed: out[b,o,f,w] = sum_i x[b,i,f,w]*ew[o,i] + eb[o]   (CIN=2, C=8)
// ---------------------------------------------------------------------------
__global__ __launch_bounds__(256) void k_embed(
    const float* __restrict__ x, const float* __restrict__ ew,
    const float* __restrict__ eb, float* __restrict__ out)
{
  long long idx = (long long)blockIdx.x * 256 + threadIdx.x;   // over B*F*W
  const long long n = (long long)B_ * F_ * W_;
  if (idx >= n) return;
  const int w = (int)(idx % W_);
  const int f = (int)((idx / W_) % F_);
  const int b = (int)(idx / ((long long)W_ * F_));
  const float x0 = x[(((long long)b * CIN_ + 0) * F_ + f) * W_ + w];
  const float x1 = x[(((long long)b * CIN_ + 1) * F_ + f) * W_ + w];
#pragma unroll
  for (int o = 0; o < C_; ++o) {
    out[(((long long)b * C_ + o) * F_ + f) * W_ + w] =
        x0 * ew[o * CIN_ + 0] + x1 * ew[o * CIN_ + 1] + eb[o];
  }
}

// ---------------------------------------------------------------------------
// frame_norm over F per (b,c,w), emitting bf16 TRANSPOSED: outT[b,c,w,f].
// Block = 32 w-columns; pass1 stats (coalesced over w), pass2 normalize with
// 32x32 LDS transpose so stores are f-contiguous.
// ---------------------------------------------------------------------------
__global__ __launch_bounds__(256) void k_frame_norm_to_bf16T(
    const float* __restrict__ X, const float* __restrict__ gamma,
    const float* __restrict__ beta, unsigned short* __restrict__ outT)
{
  __shared__ float rs[8][32], rss[8][32];
  __shared__ float mu_s[32], ri_s[32];
  __shared__ float tile[32][33];
  const int wt = blockIdx.x % (W_ / 32);
  const int bc = blockIdx.x / (W_ / 32);
  const int c  = bc % C_;
  const int w0 = wt * 32;
  const int tx = threadIdx.x & 31;
  const int ty = threadIdx.x >> 5;
  const long long xbase = (long long)bc * F_ * W_;

  float s = 0.f, ss = 0.f;
  for (int f = ty; f < F_; f += 8) {
    const float v = X[xbase + (long long)f * W_ + w0 + tx];
    s += v; ss += v * v;
  }
  rs[ty][tx] = s; rss[ty][tx] = ss;
  __syncthreads();
  if (ty == 0) {
    float a = 0.f, b2 = 0.f;
#pragma unroll
    for (int i = 0; i < 8; ++i) { a += rs[i][tx]; b2 += rss[i][tx]; }
    const float mu  = a * (1.0f / F_);
    const float var = b2 * (1.0f / F_) - mu * mu;
    mu_s[tx] = mu;
    ri_s[tx] = rsqrtf(var + 1e-5f);
  }
  __syncthreads();
  const float mu = mu_s[tx];
  const float ri = ri_s[tx];
  for (int f0 = 0; f0 < F_; f0 += 32) {
#pragma unroll
    for (int i = 0; i < 4; ++i) {
      const int fl = ty + i * 8;
      const int f  = f0 + fl;
      const float v = X[xbase + (long long)f * W_ + w0 + tx];
      tile[fl][tx] = (v - mu) * ri * gamma[c * F_ + f] + beta[c * F_ + f];
    }
    __syncthreads();
#pragma unroll
    for (int i = 0; i < 4; ++i) {
      const int wl = ty + i * 8;
      outT[((long long)bc * W_ + w0 + wl) * F_ + f0 + tx] = f2bf(tile[tx][wl]);
    }
    __syncthreads();
  }
}

// ---------------------------------------------------------------------------
// depthwise 1x3 conv (pad 1 along w) + bias + rotary + head split + scale,
// p[b,c,g,w] -> out[b,c,h,w,dh]  (transpose g<->w via LDS tile per head).
// Rotary pairs are adjacent g rows: dh=2j,2j+1.
// ---------------------------------------------------------------------------
__global__ __launch_bounds__(256) void k_conv_rot(
    const unsigned short* __restrict__ p, const float* __restrict__ cw,
    const float* __restrict__ cb, unsigned short* __restrict__ out, float scale)
{
  __shared__ float tile[DH_][33];
  const int bch = blockIdx.x;            // (b*C + c)*H + h
  const int h   = bch % H_;
  const int bc  = bch / H_;
  const int c   = bc % C_;
  const int w0  = blockIdx.y * 32;
  const int tx  = threadIdx.x & 31;      // w-local
  const int ty  = threadIdx.x >> 5;      // 8 rows of workers
  const int w   = w0 + tx;
  const float k0 = cw[c * 3 + 0], k1 = cw[c * 3 + 1], k2 = cw[c * 3 + 2];
  const float bias = cb[c];
  const long long pbase = ((long long)bc * F_ + h * DH_) * W_;

#pragma unroll
  for (int i = 0; i < 8; ++i) {
    const int jp = i * 8 + ty;           // rotary pair index 0..63
    const unsigned short* r0 = p + pbase + (long long)(2 * jp) * W_;
    const unsigned short* r1 = r0 + W_;
    const float a0 = (w > 0)      ? bf2f(r0[w - 1]) : 0.f;
    const float b0 = bf2f(r0[w]);
    const float c0 = (w < W_ - 1) ? bf2f(r0[w + 1]) : 0.f;
    const float a1 = (w > 0)      ? bf2f(r1[w - 1]) : 0.f;
    const float b1 = bf2f(r1[w]);
    const float c1 = (w < W_ - 1) ? bf2f(r1[w + 1]) : 0.f;
    const float y0 = k0 * a0 + k1 * b0 + k2 * c0 + bias;
    const float y1 = k0 * a1 + k1 * b1 + k2 * c1 + bias;
    const float inv = __expf(-0.14391156f * (float)jp);  // 10000^(-2j/128)
    const float ang = (float)w * inv;
    const float cs = __cosf(ang), sn = __sinf(ang);
    tile[2 * jp][tx]     = (y0 * cs - y1 * sn) * scale;
    tile[2 * jp + 1][tx] = (y1 * cs + y0 * sn) * scale;
  }
  __syncthreads();
#pragma unroll
  for (int i = 0; i < 4; ++i) {
    const int wl = ty + i * 8;
#pragma unroll
    for (int jj = 0; jj < 4; ++jj) {
      const int dh = tx + jj * 32;
      out[((long long)bch * W_ + w0 + wl) * DH_ + dh] = f2bf(tile[dh][wl]);
    }
  }
}

// ---------------------------------------------------------------------------
// v path: conv+bias only. Layout [b,c,g,w] with g=(h,dh) is already the
// vT[b,c,h,dh,w] layout needed by the AV GEMM -> pure elementwise conv.
// ---------------------------------------------------------------------------
__global__ __launch_bounds__(256) void k_conv_v(
    const unsigned short* __restrict__ p, const float* __restrict__ cw,
    const float* __restrict__ cb, unsigned short* __restrict__ out)
{
  long long idx = (long long)blockIdx.x * 256 + threadIdx.x;
  const long long n = (long long)B_ * C_ * F_ * W_;
  if (idx >= n) return;
  const int w = (int)(idx & (W_ - 1));
  const long long r = idx >> 9;
  const int c = (int)((r >> 10) & (C_ - 1));
  const unsigned short* row = p + r * W_;
  const float a = (w > 0)      ? bf2f(row[w - 1]) : 0.f;
  const float b = bf2f(row[w]);
  const float d = (w < W_ - 1) ? bf2f(row[w + 1]) : 0.f;
  out[idx] = f2bf(cw[c * 3] * a + cw[c * 3 + 1] * b + cw[c * 3 + 2] * d + cb[c]);
}

// ---------------------------------------------------------------------------
// softmax over the 512 keys of one row; writes bf16 attn IN PLACE over the
// fp32 logits buffer (attn row stride 1024 ushorts == logits row stride, so
// rows never overlap; all reads complete before any write).
// ---------------------------------------------------------------------------
__global__ __launch_bounds__(128) void k_softmax(
    float* logits, unsigned short* attn)
{
  __shared__ float red[128];
  const long long row = blockIdx.x;
  const float* lr = logits + row * W_;
  const int tx = threadIdx.x;
  float v[4];
#pragma unroll
  for (int i = 0; i < 4; ++i) v[i] = lr[tx + i * 128];
  float mx = fmaxf(fmaxf(v[0], v[1]), fmaxf(v[2], v[3]));
  red[tx] = mx; __syncthreads();
  for (int s2 = 64; s2 > 0; s2 >>= 1) {
    if (tx < s2) red[tx] = fmaxf(red[tx], red[tx + s2]);
    __syncthreads();
  }
  mx = red[0]; __syncthreads();
  float e[4], s = 0.f;
#pragma unroll
  for (int i = 0; i < 4; ++i) { e[i] = __expf(v[i] - mx); s += e[i]; }
  red[tx] = s; __syncthreads();
  for (int s2 = 64; s2 > 0; s2 >>= 1) {
    if (tx < s2) red[tx] += red[tx + s2];
    __syncthreads();
  }
  const float rinv = 1.0f / red[0];
  unsigned short* ar = attn + row * 1024;
#pragma unroll
  for (int i = 0; i < 4; ++i) ar[tx + i * 128] = f2bf(e[i] * rinv);
}

// ---------------------------------------------------------------------------
// lin1_dw channel mix (8x8) + relu^2:  h2[b,d,w,g] = relu(sum_c dw[d,c]*h)^2
// ---------------------------------------------------------------------------
__global__ __launch_bounds__(256) void k_mix(
    const unsigned short* __restrict__ h, const float* __restrict__ dw,
    unsigned short* __restrict__ h2)
{
  long long idx = (long long)blockIdx.x * 256 + threadIdx.x;  // over B*W*FE
  const long long n = (long long)B_ * W_ * FE_;
  if (idx >= n) return;
  const long long wg = idx % ((long long)W_ * FE_);
  const int b = (int)(idx / ((long long)W_ * FE_));
  float hv[C_];
#pragma unroll
  for (int c = 0; c < C_; ++c)
    hv[c] = bf2f(h[(long long)(b * C_ + c) * W_ * FE_ + wg]);
#pragma unroll
  for (int d = 0; d < C_; ++d) {
    float s = 0.f;
#pragma unroll
    for (int c = 0; c < C_; ++c) s += dw[d * C_ + c] * hv[c];
    const float r = fmaxf(s, 0.f);
    h2[(long long)(b * C_ + d) * W_ * FE_ + wg] = f2bf(r * r);
  }
}

// ---------------------------------------------------------------------------
extern "C" void kernel_launch(void* const* d_in, const int* in_sizes, int n_in,
                              void* d_out, int out_size, void* d_ws, size_t ws_size,
                              hipStream_t stream)
{
  (void)in_sizes; (void)n_in; (void)out_size; (void)ws_size;
  const float* x_in    = (const float*)d_in[0];
  const float* embed_w = (const float*)d_in[1];
  const float* embed_b = (const float*)d_in[2];
  const float* n1w     = (const float*)d_in[3];
  const float* n1b     = (const float*)d_in[4];
  const float* q_pw    = (const float*)d_in[5];
  const float* q_cw    = (const float*)d_in[6];
  const float* q_cb    = (const float*)d_in[7];
  const float* k_pw    = (const float*)d_in[8];
  const float* k_cw    = (const float*)d_in[9];
  const float* k_cb    = (const float*)d_in[10];
  const float* v_pw    = (const float*)d_in[11];
  const float* v_cw    = (const float*)d_in[12];
  const float* v_cb    = (const float*)d_in[13];
  const float* out_pw  = (const float*)d_in[14];
  const float* n2w     = (const float*)d_in[15];
  const float* n2b     = (const float*)d_in[16];
  const float* l1_pw   = (const float*)d_in[17];
  const float* l1_dw   = (const float*)d_in[18];
  const float* l2_pw   = (const float*)d_in[19];
  float* out = (float*)d_out;

  char* ws = (char*)d_ws;
  const size_t MB = 1ull << 20;
  float*          xbuf  = (float*)(ws + 0);
  unsigned short* xnT   = (unsigned short*)(ws + 64 * MB);
  unsigned short* Wq    = (unsigned short*)(ws + 96 * MB);
  unsigned short* Wk    = (unsigned short*)(ws + 112 * MB);
  unsigned short* Wv    = (unsigned short*)(ws + 128 * MB);
  unsigned short* Wo    = (unsigned short*)(ws + 144 * MB);
  unsigned short* Wl1   = (unsigned short*)(ws + 160 * MB);
  unsigned short* Wl2   = (unsigned short*)(ws + 224 * MB);
  unsigned short* pq    = (unsigned short*)(ws + 288 * MB);
  unsigned short* pk    = (unsigned short*)(ws + 320 * MB);
  unsigned short* pv    = (unsigned short*)(ws + 352 * MB);
  unsigned short* qb    = (unsigned short*)(ws + 384 * MB);
  unsigned short* kb    = (unsigned short*)(ws + 416 * MB);
  unsigned short* vT    = (unsigned short*)(ws + 448 * MB);
  float*          logits= (float*)(ws + 480 * MB);
  unsigned short* attn  = (unsigned short*)(ws + 480 * MB);   // in-place alias
  unsigned short* aT    = (unsigned short*)(ws + 288 * MB);   // reuse pq
  unsigned short* hbuf  = (unsigned short*)(ws + 480 * MB);   // reuse logits
  unsigned short* h2    = (unsigned short*)(ws + 608 * MB);

  // ---- stage weights to bf16 ----
  auto cvt = [&](const float* src, unsigned short* dst, long long nelem) {
    const long long n4 = nelem / 4;
    k_f32_to_bf16<<<(unsigned)((n4 + 255) / 256), 256, 0, stream>>>(src, dst, n4);
  };
  cvt(q_pw,  Wq,  (long long)C_ * F_ * F_);
  cvt(k_pw,  Wk,  (long long)C_ * F_ * F_);
  cvt(v_pw,  Wv,  (long long)C_ * F_ * F_);
  cvt(out_pw, Wo, (long long)C_ * F_ * F_);
  cvt(l1_pw, Wl1, (long long)C_ * FE_ * F_);
  cvt(l2_pw, Wl2, (long long)C_ * F_ * FE_);

  // ---- embed + norm1 ----
  {
    const long long n = (long long)B_ * F_ * W_;
    k_embed<<<(unsigned)((n + 255) / 256), 256, 0, stream>>>(x_in, embed_w, embed_b, xbuf);
  }
  k_frame_norm_to_bf16T<<<B_ * C_ * (W_ / 32), 256, 0, stream>>>(xbuf, n1w, n1b, xnT);

  // ---- q/k/v projections: D[g][w] = Wt(gxf) x xn(fxw), per (b,c) ----
  const dim3 gproj(F_ / 128, W_ / 128, B_ * C_);   // (8,4,32)
  gemm_bf16_wmma<2><<<gproj, 128, 0, stream>>>(Wq, xnT, nullptr, pq, nullptr,
      F_, F_, W_, F_,
      C_, 0ll, (long long)F_ * F_,
      1, (long long)W_ * F_, 0ll,
      1, (long long)F_ * W_, 0ll);
  gemm_bf16_wmma<2><<<gproj, 128, 0, stream>>>(Wk, xnT, nullptr, pk, nullptr,
      F_, F_, W_, F_,
      C_, 0ll, (long long)F_ * F_,
      1, (long long)W_ * F_, 0ll,
      1, (long long)F_ * W_, 0ll);
  gemm_bf16_wmma<2><<<gproj, 128, 0, stream>>>(Wv, xnT, nullptr, pv, nullptr,
      F_, F_, W_, F_,
      C_, 0ll, (long long)F_ * F_,
      1, (long long)W_ * F_, 0ll,
      1, (long long)F_ * W_, 0ll);

  // ---- dwconv (+rotary for q/k; q gets 1/sqrt(F) folded in) ----
  const dim3 gconv(B_ * C_ * H_, W_ / 32);
  k_conv_rot<<<gconv, 256, 0, stream>>>(pq, q_cw, q_cb, qb, 1.0f / 32.0f);
  k_conv_rot<<<gconv, 256, 0, stream>>>(pk, k_cw, k_cb, kb, 1.0f);
  {
    const long long n = (long long)B_ * C_ * F_ * W_;
    k_conv_v<<<(unsigned)((n + 255) / 256), 256, 0, stream>>>(pv, v_cw, v_cb, vT);
  }

  // ---- attention: logits = q x k^T  (scale already in q) ----
  const dim3 gqk(W_ / 128, W_ / 128, B_ * C_ * H_);  // (4,4,256)
  gemm_bf16_wmma<0><<<gqk, 128, 0, stream>>>(qb, kb, logits, nullptr, nullptr,
      DH_, DH_, W_, DH_,
      1, (long long)W_ * DH_, 0ll,
      1, (long long)W_ * DH_, 0ll,
      1, (long long)W_ * W_, 0ll);

  k_softmax<<<B_ * C_ * H_ * W_, 128, 0, stream>>>(logits, attn);

  // ---- a = attn x v  -> aT[b,c,w, h*128+dh] bf16 ----
  const dim3 gav(W_ / 128, DH_ / 128, B_ * C_ * H_); // (4,1,256)
  gemm_bf16_wmma<2><<<gav, 128, 0, stream>>>(attn, vT, nullptr, aT, nullptr,
      1024, W_, F_, W_,
      1, (long long)W_ * 1024, 0ll,
      1, (long long)DH_ * W_, 0ll,
      H_, (long long)W_ * F_, (long long)DH_);

  // ---- out projection with residual, in place on x ----
  gemm_bf16_wmma<1><<<gproj, 128, 0, stream>>>(Wo, aT, xbuf, nullptr, xbuf,
      F_, F_, W_, F_,
      C_, 0ll, (long long)F_ * F_,
      1, (long long)W_ * F_, 0ll,
      1, (long long)F_ * W_, 0ll);

  // ---- norm2 -> xn2T (reuse xnT) ----
  k_frame_norm_to_bf16T<<<B_ * C_ * (W_ / 32), 256, 0, stream>>>(xbuf, n2w, n2b, xnT);

  // ---- lin1: D[w][g] = xn2T(wxf) x Wl1^T -> h[b,c,w,g] bf16 ----
  const dim3 gl1(W_ / 128, FE_ / 128, B_ * C_);      // (4,32,32)
  gemm_bf16_wmma<2><<<gl1, 128, 0, stream>>>(xnT, Wl1, nullptr, hbuf, nullptr,
      F_, F_, FE_, F_,
      1, (long long)W_ * F_, 0ll,
      C_, 0ll, (long long)FE_ * F_,
      1, (long long)W_ * FE_, 0ll);

  // ---- channel mix + relu^2 ----
  {
    const long long n = (long long)B_ * W_ * FE_;
    k_mix<<<(unsigned)((n + 255) / 256), 256, 0, stream>>>(hbuf, l1_dw, h2);
  }

  // ---- lin2 + residual -> d_out fp32 ----
  gemm_bf16_wmma<1><<<gproj, 128, 0, stream>>>(Wl2, h2, out, nullptr, xbuf,
      FE_, FE_, W_, FE_,
      C_, 0ll, (long long)F_ * FE_,
      1, (long long)W_ * FE_, 0ll,
      1, (long long)F_ * W_, 0ll);
}
